// DINN_5961414606884
// MI455X (gfx1250) — matrix operand order repaired and checked by reading, not compiled, not run on hardware
//
#include <hip/hip_runtime.h>
#include <hip/hip_bf16.h>

typedef __attribute__((ext_vector_type(16))) __bf16 v16bf;
typedef __attribute__((ext_vector_type(8)))  float  v8f;

static constexpr int kD     = 512;
static constexpr int kBatch = 16384;
static constexpr int kM     = 64;    // rows per workgroup
static constexpr int kPitch = 520;   // LDS row pitch in bf16 elems (512 + 8 pad -> 4-bank skew)

static __device__ __forceinline__ unsigned short f2bf(float f) {
    __bf16 h = (__bf16)f;
    return __builtin_bit_cast(unsigned short, h);
}
static __device__ __forceinline__ float bf2f(unsigned short u) {
    __bf16 h = __builtin_bit_cast(__bf16, u);
    return (float)h;
}

// ---------------------------------------------------------------------------
// Prep: build dense U^T from iw (strict upper triangle, row-major i<j) and
// pre-split W and U^T into bf16 hi/lo planes so the GEMM kernel loads bf16
// fragments directly. One thread per (i,j) element of the 512x512 matrices.
// ---------------------------------------------------------------------------
__global__ void dinn_prep(const float* __restrict__ W, const float* __restrict__ iw,
                          unsigned short* __restrict__ Whi, unsigned short* __restrict__ Wlo,
                          unsigned short* __restrict__ UThi, unsigned short* __restrict__ UTlo) {
    int idx = blockIdx.x * blockDim.x + threadIdx.x;   // 0 .. 512*512-1
    int i = idx >> 9;        // row
    int j = idx & (kD - 1);  // col

    float w = W[idx];
    unsigned short wh = f2bf(w);
    Whi[idx] = wh;
    Wlo[idx] = f2bf(w - bf2f(wh));

    // UT[i][j] = U[j][i]; U[a][b] nonzero iff a < b, flat index a*(D-1)-a(a-1)/2+(b-a-1)
    float u = 0.0f;
    if (j < i) {
        int p = j * (kD - 1) - (j * (j - 1)) / 2 + (i - j - 1);
        u = iw[p];
    }
    unsigned short uh = f2bf(u);
    UThi[idx] = uh;
    UTlo[idx] = f2bf(u - bf2f(uh));
}

// ---------------------------------------------------------------------------
// Main fused kernel: per-WG 64 rows, 8 waves = {W,U} x 4 column-quarters.
// bf16 hi/lo split GEMM on v_wmma_f32_16x16x32_bf16, fused sigmoid epilogue,
// shfl + ds_add_f32 row reduction, final sigmoid store.
// ---------------------------------------------------------------------------
__global__ __launch_bounds__(256, 1)
void dinn_main(const float* __restrict__ x, const float* __restrict__ bvec,
               const unsigned short* __restrict__ Whi, const unsigned short* __restrict__ Wlo,
               const unsigned short* __restrict__ UThi, const unsigned short* __restrict__ UTlo,
               float* __restrict__ out) {
    __shared__ unsigned short lds_xhi[kM * kPitch];
    __shared__ unsigned short lds_xlo[kM * kPitch];
    __shared__ float lds_b[kD];
    __shared__ float lds_rowsum[kM];

    const int tid     = threadIdx.x;
    const int rowbase = blockIdx.x * kM;

    if (tid < kM) lds_rowsum[tid] = 0.0f;
    lds_b[tid]       = bvec[tid];
    lds_b[tid + 256] = bvec[tid + 256];

    // Stage x tile [64 x 512] f32 -> bf16 hi/lo in LDS (coalesced float4 loads).
    {
        const float4* xg = (const float4*)(x + rowbase * kD);
        #pragma unroll 4
        for (int it = 0; it < (kM * kD / 4) / 256; ++it) {
            int e4 = tid + it * 256;
            float4 v = xg[e4];
            int e   = e4 * 4;
            int row = e >> 9;
            int k   = e & (kD - 1);
            int off = row * kPitch + k;
            float f0 = v.x, f1 = v.y, f2 = v.z, f3 = v.w;
            unsigned short h0 = f2bf(f0), h1 = f2bf(f1), h2 = f2bf(f2), h3 = f2bf(f3);
            unsigned short l0 = f2bf(f0 - bf2f(h0)), l1 = f2bf(f1 - bf2f(h1));
            unsigned short l2 = f2bf(f2 - bf2f(h2)), l3 = f2bf(f3 - bf2f(h3));
            uint2 hp, lp;
            hp.x = (unsigned)h0 | ((unsigned)h1 << 16);
            hp.y = (unsigned)h2 | ((unsigned)h3 << 16);
            lp.x = (unsigned)l0 | ((unsigned)l1 << 16);
            lp.y = (unsigned)l2 | ((unsigned)l3 << 16);
            *(uint2*)(lds_xhi + off) = hp;
            *(uint2*)(lds_xlo + off) = lp;
        }
    }
    __syncthreads();

    const int lane     = tid & 31;
    const int wave     = tid >> 5;
    const int lane16   = lane & 15;
    const int laneHalf = lane >> 4;
    const int mat      = wave >> 2;   // 0 = gates (W), 1 = quad (U^T)
    const int nq       = wave & 3;    // column quarter

    const unsigned short* Bhi = mat ? UThi : Whi;
    const unsigned short* Blo = mat ? UTlo : Wlo;

    float rowpart[4][8];
    #pragma unroll
    for (int s = 0; s < 4; ++s)
        #pragma unroll
        for (int r = 0; r < 8; ++r) rowpart[s][r] = 0.0f;

    #pragma unroll 1
    for (int nt = 0; nt < 8; ++nt) {
        const int n0  = nq * 128 + nt * 16;
        const int col = n0 + lane16;

        v8f acc[4];
        #pragma unroll
        for (int s = 0; s < 4; ++s) acc[s] = (v8f){};

        const unsigned short* bph = Bhi + col * kD + laneHalf * 16;
        const unsigned short* bpl = Blo + col * kD + laneHalf * 16;

        // unroll 2: two independent k-steps let the scheduler fill the
        // WMMA->VALU hazard slots (previously v_nop) with the other step's
        // ds_load/wmma work. Full unroll caused >256-VGPR spills (round 1);
        // unroll 2 adds only ~1 fragment set (~+32 VGPRs).
        #pragma unroll 2
        for (int k0 = 0; k0 < kD; k0 += 32) {
            // Prefetch next k-step's B rows into cache (global_prefetch_b8).
            __builtin_prefetch(bph + k0 + 32, 0, 3);
            __builtin_prefetch(bpl + k0 + 32, 0, 3);

            // B fragment (32x16): lane holds 16 contiguous K values; lane half
            // selects K 0-15 vs 16-31.
            union { v16bf v; uint4 q[2]; } bh, bl;
            const uint4* ph = (const uint4*)(bph + k0);
            const uint4* pl = (const uint4*)(bpl + k0);
            bh.q[0] = ph[0]; bh.q[1] = ph[1];
            bl.q[0] = pl[0]; bl.q[1] = pl[1];

            #pragma unroll
            for (int s = 0; s < 4; ++s) {
                // A fragment (16x32): lane = row; 16 contiguous K at
                // k0 + laneHalf*16.
                const int base = (s * 16 + lane16) * kPitch + k0 + laneHalf * 16;
                union { v16bf v; uint4 q[2]; } ah, al;
                ah.q[0] = *(const uint4*)(lds_xhi + base);
                ah.q[1] = *(const uint4*)(lds_xhi + base + 8);
                al.q[0] = *(const uint4*)(lds_xlo + base);
                al.q[1] = *(const uint4*)(lds_xlo + base + 8);

                acc[s] = __builtin_amdgcn_wmma_f32_16x16x32_bf16(
                    false, ah.v, false, bh.v, (short)0, acc[s], false, false);
                acc[s] = __builtin_amdgcn_wmma_f32_16x16x32_bf16(
                    false, ah.v, false, bl.v, (short)0, acc[s], false, false);
                acc[s] = __builtin_amdgcn_wmma_f32_16x16x32_bf16(
                    false, al.v, false, bh.v, (short)0, acc[s], false, false);
            }
        }

        // Fused epilogue: C layout -> lane N = lane16, rows M = r + 8*laneHalf.
        const float bias = lds_b[col];
        #pragma unroll
        for (int s = 0; s < 4; ++s) {
            const float* xcol = x + (rowbase + s * 16 + laneHalf * 8) * kD + col;
            #pragma unroll
            for (int r = 0; r < 8; ++r) {
                float xv = xcol[r * kD];
                float a  = acc[s][r];
                float c;
                if (mat == 0) {
                    float gate = 1.0f / (1.0f + __expf(-(a + bias)));
                    c = gate * xv;
                } else {
                    c = a * xv;
                }
                rowpart[s][r] += c;
            }
        }
    }

    // Reduce each row's 16 column-partials within the lane half, then
    // accumulate across waves via LDS float atomics.
    #pragma unroll
    for (int s = 0; s < 4; ++s) {
        #pragma unroll
        for (int r = 0; r < 8; ++r) {
            float v = rowpart[s][r];
            v += __shfl_xor(v, 1, 32);
            v += __shfl_xor(v, 2, 32);
            v += __shfl_xor(v, 4, 32);
            v += __shfl_xor(v, 8, 32);
            if (lane16 == 0)
                atomicAdd(&lds_rowsum[s * 16 + laneHalf * 8 + r], v);
        }
    }
    __syncthreads();

    if (tid < kM) {
        float s = lds_rowsum[tid];
        out[rowbase + tid] = 1.0f / (1.0f + __expf(-s));
    }
}

// ---------------------------------------------------------------------------
extern "C" void kernel_launch(void* const* d_in, const int* in_sizes, int n_in,
                              void* d_out, int out_size, void* d_ws, size_t ws_size,
                              hipStream_t stream) {
    const float* x  = (const float*)d_in[0];   // [16384, 512]
    const float* W  = (const float*)d_in[1];   // [512, 512]
    const float* b  = (const float*)d_in[2];   // [512]
    const float* iw = (const float*)d_in[3];   // [130816]
    float* out = (float*)d_out;                // [16384, 1]

    unsigned short* Whi  = (unsigned short*)d_ws;      // 512KB each plane
    unsigned short* Wlo  = Whi  + kD * kD;
    unsigned short* UThi = Wlo  + kD * kD;
    unsigned short* UTlo = UThi + kD * kD;             // total 2 MB scratch

    dinn_prep<<<(kD * kD) / 256, 256, 0, stream>>>(W, iw, Whi, Wlo, UThi, UTlo);
    dinn_main<<<kBatch / kM, 256, 0, stream>>>(x, b, Whi, Wlo, UThi, UTlo, out);
}
